// sLSTM_70772471103948
// MI455X (gfx1250) — matrix operand order repaired and checked by reading, not compiled, not run on hardware
//
#include <hip/hip_runtime.h>

// ---------------- problem constants ----------------
#define B_ 64
#define T_ 1024
#define I_ 128
#define H_ 512
#define NG_ 2048   // 4*H

typedef __attribute__((ext_vector_type(8)))  float  v8f;
typedef __attribute__((ext_vector_type(4)))  float  v4f;
typedef __attribute__((ext_vector_type(16))) __bf16 v16bf;

// float -> bf16 round-to-nearest-even (bit-level)
__device__ __forceinline__ __bf16 f2bf(float f) {
    unsigned u = __float_as_uint(f);
    unsigned r = (u + 0x7FFFu + ((u >> 16) & 1u)) >> 16;
    unsigned short s = (unsigned short)r;
    return *reinterpret_cast<__bf16*>(&s);
}

__device__ __forceinline__ float sigmoidf_(float x) {
    return 1.0f / (1.0f + __expf(-x));
}

// WGP-scope prefetch (scope field 0): fills WGP$ + GL2 on miss.
// clang's __builtin_prefetch only produces SE/DEV/SYS scopes, so use inline asm.
__device__ __forceinline__ void prefetch_wgp(const void* p) {
    asm volatile("global_prefetch_b8 %0, off" :: "v"(p));
}

// ---- WMMA bf16 fragment loads (per ISA 16-bit layouts, wave32) ----
// A (16x32, MxK): lane m=l&15, half=l>>4; K runs [half*8,+8) and [16+half*8,+8)
__device__ __forceinline__ v16bf load_a_frag(const __bf16* rp, int half) {
    union { v4f q[2]; v16bf v; } u;
    u.q[0] = *reinterpret_cast<const v4f*>(rp + half * 8);
    u.q[1] = *reinterpret_cast<const v4f*>(rp + 16 + half * 8);
    return u.v;
}
// B (32x16, KxN): lane n=l&15, half=l>>4; K runs [half*8,+8) and [16+half*8,+8)
// column n of B == row n of row-major W[N][K] -> no repack needed.
__device__ __forceinline__ v16bf load_b_frag(const __bf16* rp, int half) {
    union { v4f q[2]; v16bf v; } u;
    u.q[0] = *reinterpret_cast<const v4f*>(rp + half * 8);
    u.q[1] = *reinterpret_cast<const v4f*>(rp + 16 + half * 8);
    return u.v;
}

// device-scope sense barrier among the 16 WGs of one batch group.
// Asymmetric fences: release (global_wb) before arrival, acquire (global_inv) after.
__device__ __forceinline__ void group_barrier(unsigned* bar) {
    __builtin_amdgcn_fence(__ATOMIC_RELEASE, "agent");
    __syncthreads();
    if (threadIdx.x == 0) {
        unsigned gen  = __hip_atomic_load(bar + 1, __ATOMIC_RELAXED, __HIP_MEMORY_SCOPE_AGENT);
        unsigned prev = __hip_atomic_fetch_add(bar, 1u, __ATOMIC_ACQ_REL, __HIP_MEMORY_SCOPE_AGENT);
        if (prev == 15u) {
            __hip_atomic_store(bar, 0u, __ATOMIC_RELAXED, __HIP_MEMORY_SCOPE_AGENT);
            __hip_atomic_fetch_add(bar + 1, 1u, __ATOMIC_RELEASE, __HIP_MEMORY_SCOPE_AGENT);
        } else {
            while (__hip_atomic_load(bar + 1, __ATOMIC_RELAXED, __HIP_MEMORY_SCOPE_AGENT) == gen) {
                __builtin_amdgcn_s_sleep(1);
            }
        }
    }
    __syncthreads();
    __builtin_amdgcn_fence(__ATOMIC_ACQUIRE, "agent");
}

// ---------------- prep: bf16 conversions + state init ----------------
__global__ void slstm_prep(const float* __restrict__ x,
                           const float* __restrict__ Wih,
                           const float* __restrict__ Whh,
                           const float* __restrict__ bih,
                           const float* __restrict__ bhh,
                           const float* __restrict__ Wf,
                           __bf16* __restrict__ xbf,
                           __bf16* __restrict__ wihb,
                           __bf16* __restrict__ whhb,
                           __bf16* __restrict__ wfb,
                           float*  __restrict__ biasg,
                           __bf16* __restrict__ hbuf,
                           unsigned* __restrict__ bars) {
    size_t tid    = (size_t)blockIdx.x * blockDim.x + threadIdx.x;
    size_t stride = (size_t)gridDim.x * blockDim.x;
    for (size_t k = tid; k < (size_t)B_ * T_ * I_; k += stride) xbf[k]  = f2bf(x[k]);
    for (size_t k = tid; k < (size_t)NG_ * I_;     k += stride) wihb[k] = f2bf(Wih[k]);
    for (size_t k = tid; k < (size_t)NG_ * H_;     k += stride) whhb[k] = f2bf(Whh[k]);
    for (size_t k = tid; k < (size_t)H_ * H_;      k += stride) wfb[k]  = f2bf(Wf[k]);
    for (size_t k = tid; k < (size_t)NG_;          k += stride) biasg[k] = bih[k] + bhh[k];
    for (size_t k = tid; k < (size_t)2 * B_ * H_;  k += stride) hbuf[k] = f2bf(0.0f);
    for (size_t k = tid; k < 16;                   k += stride) bars[k] = 0u;
}

// ---------------- fused persistent sLSTM scan ----------------
// grid = 64 WGs: blockIdx = bg*16 + ns ; bg = batch group (16 rows), ns = hidden slice (32 cols)
// block = 512 threads = 16 waves.
// Gate GEMM: 8 tiles (i/f/g/o x 2 sub-tiles) x 2 K-groups -> wave w: tile tt=w&7, kgroup kg=w>>3.
//   kg0: x_t@W_ih (4 kblocks) + h@W_hh kblocks 0..5 (+bias); kg1: h@W_hh kblocks 6..15.
// fe GEMM: 2 tiles x 8 K-groups -> wave w: tile sub=w&1, kgroup kf=w>>1 (2 kblocks each).
// Partials reduced in LDS during the elementwise phases.
__global__ __launch_bounds__(512) void slstm_scan(
        const __bf16* __restrict__ xbf,    // [B][T][I]
        const __bf16* __restrict__ Wih,    // [NG][I]
        const __bf16* __restrict__ Whh,    // [NG][H]
        const __bf16* __restrict__ Wfb,    // [H][H]
        const float*  __restrict__ biasg,  // [NG] = b_ih + b_hh
        const float*  __restrict__ bfv,    // [H]
        __bf16*       __restrict__ hbuf,   // [2][B][H] double buffer
        unsigned*     __restrict__ bars,   // [4][2]
        float*        __restrict__ out,    // [B][T][H]
        float*        __restrict__ outh,   // [B][H]
        float*        __restrict__ outc)   // [B][H]
{
    const int tid  = threadIdx.x;
    const int lane = tid & 31;
    const int wv   = tid >> 5;            // 0..15
    const int bg   = blockIdx.x >> 4;
    const int ns   = blockIdx.x & 15;
    const int half = lane >> 4;
    const int nn   = lane & 15;

    // 16 KB LDS: gate partials [2 kg][16 m][128 col] == fe partials [8 kf][16 m][32 col]
    __shared__ float ldsP[4096];

    // gate-tile mapping
    const int tt   = wv & 7;
    const int kg   = wv >> 3;             // 0 or 1
    const int gt   = tt >> 1;
    const int gsub = tt & 1;
    const int gcol = gt * H_ + ns * 32 + gsub * 16 + nn;  // W_ih/W_hh row for this lane's B column

    // fe mapping
    const int fsub = wv & 1;
    const int kf   = wv >> 1;             // 0..7
    const int fcol = ns * 32 + fsub * 16 + nn;            // Wf row (always < H)

    const int arow = bg * 16 + nn;                        // A-matrix batch row for this lane
    const __bf16* xrow_base = xbf + (size_t)arow * T_ * I_;
    const __bf16* wih_row   = Wih + (size_t)gcol * I_;
    const __bf16* whh_row   = Whh + (size_t)gcol * H_;
    const __bf16* wf_row    = Wfb + (size_t)fcol * H_;

    const float bias_val = (kg == 0) ? biasg[gcol] : 0.0f;

    // elementwise mapping: one (m, j) element per thread
    const int em  = tid >> 5;             // 0..15 (row)
    const int ej  = tid & 31;             // 0..31 (hidden col within slice)
    const int ecol = ns * 32 + ej;
    const float bf_t = bfv[ecol];
    float cst = 0.0f, hreg = 0.0f;

    unsigned* bar = bars + bg * 2;

    for (int t = 0; t < T_; ++t) {
        const int p = t & 1;
        const __bf16* xrow  = xrow_base + (size_t)t * I_;
        const __bf16* hprev = hbuf + (size_t)p * B_ * H_ + (size_t)arow * H_;

        // prefetch next timestep's x rows into the WGP cache
        // (one wave's 32 lanes cover all 16 rows x 256B: half 0 -> +0B, half 1 -> +128B)
        if (wv == 0 && t + 1 < T_)
            prefetch_wgp(xrow + I_ + half * 64);

        // ---------- gates = bias + x_t@W_ih^T + h_prev@W_hh^T (K split over 2 waves) ----------
        v8f acc;
#pragma unroll
        for (int r = 0; r < 8; ++r) acc[r] = bias_val;

        if (kg == 0) {
#pragma unroll
            for (int kb = 0; kb < 4; ++kb) {
                v16bf a = load_a_frag(xrow + kb * 32, half);
                v16bf b = load_b_frag(wih_row + kb * 32, half);
                acc = __builtin_amdgcn_wmma_f32_16x16x32_bf16(false, a, false, b, (short)0, acc, false, false);
            }
#pragma unroll
            for (int kb = 0; kb < 6; ++kb) {
                v16bf a = load_a_frag(hprev + kb * 32, half);
                v16bf b = load_b_frag(whh_row + kb * 32, half);
                acc = __builtin_amdgcn_wmma_f32_16x16x32_bf16(false, a, false, b, (short)0, acc, false, false);
            }
        } else {
#pragma unroll
            for (int kb = 6; kb < 16; ++kb) {
                v16bf a = load_a_frag(hprev + kb * 32, half);
                v16bf b = load_b_frag(whh_row + kb * 32, half);
                acc = __builtin_amdgcn_wmma_f32_16x16x32_bf16(false, a, false, b, (short)0, acc, false, false);
            }
        }
#pragma unroll
        for (int r = 0; r < 8; ++r)
            ldsP[kg * 2048 + (half * 8 + r) * 128 + tt * 16 + nn] = acc[r];
        __syncthreads();

        // ---------- cell update: reduce 2 K-partials, i/f/g/o share hidden index ----------
        {
            const float iv = sigmoidf_(ldsP[em * 128 + ej]       + ldsP[2048 + em * 128 + ej]);
            const float fv = sigmoidf_(ldsP[em * 128 + 32 + ej]  + ldsP[2048 + em * 128 + 32 + ej]);
            const float gv = tanhf    (ldsP[em * 128 + 64 + ej]  + ldsP[2048 + em * 128 + 64 + ej]);
            const float ov = sigmoidf_(ldsP[em * 128 + 96 + ej]  + ldsP[2048 + em * 128 + 96 + ej]);
            const float c1 = fv * cst + iv * gv;
            const float hv = ov * tanhf(c1);
            cst  = c1;
            hreg = hv;
            hbuf[(size_t)(1 - p) * B_ * H_ + (size_t)(bg * 16 + em) * H_ + ecol] = f2bf(hv);
            out[((size_t)(bg * 16 + em) * T_ + t) * H_ + ecol] = hv;
        }

        // ---------- publish h_new, sync the 16 WGs of this batch group ----------
        group_barrier(bar);   // includes __syncthreads: ldsP reusable after this

        // ---------- fe partials: h_new@Wf^T, K split 8 ways ----------
        {
            v8f fa;
#pragma unroll
            for (int r = 0; r < 8; ++r) fa[r] = 0.0f;
            const __bf16* hnew = hbuf + (size_t)(1 - p) * B_ * H_ + (size_t)arow * H_;
#pragma unroll
            for (int q = 0; q < 2; ++q) {
                const int kb = kf * 2 + q;
                v16bf a = load_a_frag(hnew + kb * 32, half);
                v16bf b = load_b_frag(wf_row + kb * 32, half);
                fa = __builtin_amdgcn_wmma_f32_16x16x32_bf16(false, a, false, b, (short)0, fa, false, false);
            }
#pragma unroll
            for (int r = 0; r < 8; ++r)
                ldsP[kf * 512 + (half * 8 + r) * 32 + fsub * 16 + nn] = fa[r];
        }
        __syncthreads();

        // ---------- reduce fe partials, c = exp(fe)*c ----------
        {
            float s = bf_t;
#pragma unroll
            for (int g = 0; g < 8; ++g)
                s += ldsP[g * 512 + em * 32 + ej];
            cst *= __expf(s);
        }
        __syncthreads();   // protect ldsP before next iteration's gate-partial writes
    }

    // ---------- final h, c ----------
    outh[(size_t)(bg * 16 + em) * H_ + ecol] = hreg;
    outc[(size_t)(bg * 16 + em) * H_ + ecol] = cst;
}

// ---------------- host launcher ----------------
extern "C" void kernel_launch(void* const* d_in, const int* in_sizes, int n_in,
                              void* d_out, int out_size, void* d_ws, size_t ws_size,
                              hipStream_t stream) {
    const float* x   = (const float*)d_in[0];
    const float* Wih = (const float*)d_in[1];
    const float* Whh = (const float*)d_in[2];
    const float* bih = (const float*)d_in[3];
    const float* bhh = (const float*)d_in[4];
    const float* Wf  = (const float*)d_in[5];
    const float* bfv = (const float*)d_in[6];
    // d_in[7]=Wi, d_in[8]=bi are unused by the reference computation.

    size_t off = 0;
    auto carve = [&](size_t bytes) -> void* {
        void* p = (char*)d_ws + off;
        off += (bytes + 255) & ~(size_t)255;
        return p;
    };
    __bf16*   xbf   = (__bf16*)  carve((size_t)B_ * T_ * I_ * 2);
    __bf16*   wihb  = (__bf16*)  carve((size_t)NG_ * I_ * 2);
    __bf16*   whhb  = (__bf16*)  carve((size_t)NG_ * H_ * 2);
    __bf16*   wfb   = (__bf16*)  carve((size_t)H_ * H_ * 2);
    float*    biasg = (float*)   carve((size_t)NG_ * 4);
    __bf16*   hbuf  = (__bf16*)  carve((size_t)2 * B_ * H_ * 2);
    unsigned* bars  = (unsigned*)carve(16 * sizeof(unsigned));

    float* out  = (float*)d_out;
    float* outh = out + (size_t)B_ * T_ * H_;
    float* outc = outh + (size_t)B_ * H_;

    slstm_prep<<<4096, 256, 0, stream>>>(x, Wih, Whh, bih, bhh, Wf,
                                         xbf, wihb, whhb, wfb, biasg, hbuf, bars);
    slstm_scan<<<64, 512, 0, stream>>>(xbf, wihb, whhb, wfb, biasg, bfv,
                                       hbuf, bars, out, outh, outc);
}
